// MultiHeadAttention_59365037965756
// MI455X (gfx1250) — matrix operand order repaired and checked by reading, not compiled, not run on hardware
//
#include <hip/hip_runtime.h>

typedef __bf16 bf16_t;
typedef __attribute__((ext_vector_type(16))) __bf16 v16bf;
typedef __attribute__((ext_vector_type(8)))  __bf16 v8bf;
typedef __attribute__((ext_vector_type(8)))  float  v8f;
typedef int v4i_vs __attribute__((vector_size(16)));

#define BSZ    2
#define SLEN   2048
#define DIMN   2048
#define NHEADS 16
#define DHEAD  128
#define MTOT   (BSZ * SLEN)

#define AS1 __attribute__((address_space(1)))
#define AS3 __attribute__((address_space(3)))

// ---------- scalar helpers ----------

__device__ __forceinline__ unsigned short f2bfu(float f) {
  union { float f; unsigned u; } a; a.f = f;
  unsigned r = a.u + 0x7FFFu + ((a.u >> 16) & 1u);  // round-to-nearest-even
  return (unsigned short)(r >> 16);
}
__device__ __forceinline__ bf16_t f2bf(float f) {
  unsigned short h = f2bfu(f);
  return __builtin_bit_cast(bf16_t, h);
}
__device__ __forceinline__ float bf2f(unsigned short h) {
  union { unsigned u; float f; } a; a.u = ((unsigned)h) << 16;
  return a.f;
}

__device__ __forceinline__ v8f wmma_bf16(v16bf a, v16bf b, v8f c) {
  return __builtin_amdgcn_wmma_f32_16x16x32_bf16(
      false, a, false, b, (short)0, c, false, false);
}

// A fragment (16x32, MxK) from row-major source, leading dim ld.
__device__ __forceinline__ v16bf load_a_frag(const bf16_t* __restrict__ base,
                                             int ld, int row0, int k0, int lane) {
  int r = lane & 15, hi = lane >> 4;
  const bf16_t* p = base + (size_t)(row0 + r) * ld + k0 + hi * 8;
  union { v16bf v; v8bf h[2]; } u;
  u.h[0] = *(const v8bf*)(p);
  u.h[1] = *(const v8bf*)(p + 16);
  return u.v;
}

// B fragment (32x16, KxN) from row-major [N,K] source (i.e. B^T), leading dim ld.
__device__ __forceinline__ v16bf load_b_frag(const bf16_t* __restrict__ base,
                                             int ld, int col0, int k0, int lane) {
  int n = lane & 15, hi = lane >> 4;
  const bf16_t* p = base + (size_t)(col0 + n) * ld + k0 + hi * 8;
  union { v16bf v; v8bf h[2]; } u;
  u.h[0] = *(const v8bf*)(p);
  u.h[1] = *(const v8bf*)(p + 16);
  return u.v;
}

// ---------- CDNA5 async global->LDS staging (16B per lane) ----------

__device__ __forceinline__ void async_copy16(const bf16_t* g, bf16_t* l) {
#if __has_builtin(__builtin_amdgcn_global_load_async_to_lds_b128)
  // prototype (from probe): (v4i addrspace(1)*, v4i addrspace(3)*, imm offset, imm cpol)
  __builtin_amdgcn_global_load_async_to_lds_b128(
      (AS1 v4i_vs*)g, (AS3 v4i_vs*)l, 0, 0);
#else
  *(v8bf*)l = *(const v8bf*)g;  // sync fallback: load + ds_store
#endif
}

#if __has_builtin(__builtin_amdgcn_s_wait_asynccnt)
#define WAIT_ASYNC(n) __builtin_amdgcn_s_wait_asynccnt(n)
#else
#define WAIT_ASYNC(n) asm volatile("s_wait_asynccnt %0" :: "i"(n) : "memory")
#endif

// ---------- f32 -> bf16 pack ----------

__global__ __launch_bounds__(256) void cvt_f32_bf16(const float* __restrict__ in,
                                                    bf16_t* __restrict__ out, int n) {
  int i = (blockIdx.x * 256 + threadIdx.x) * 4;
  if (i < n) {
    float4 f = *(const float4*)(in + i);
    unsigned short* op = (unsigned short*)out + i;
    ushort4 o;
    o.x = f2bfu(f.x); o.y = f2bfu(f.y); o.z = f2bfu(f.z); o.w = f2bfu(f.w);
    *(ushort4*)op = o;
  }
}

// ---------- GEMM: C[M,N] = A[M,K] @ B[N,K]^T (+ bias) ----------
// 8 waves as 4(M) x 2(N); wave tile 32x64 (8 WMMA accumulators); block 128x128.

__global__ __launch_bounds__(256) void gemm_bf16_nt(
    const bf16_t* __restrict__ A, const bf16_t* __restrict__ B,
    const float* __restrict__ bias,
    float* __restrict__ Cf, bf16_t* __restrict__ Cb,
    int M, int N, int K) {
  int lane = threadIdx.x & 31;
  int wave = threadIdx.x >> 5;
  int wm = wave & 3, wn = wave >> 2;
  int m0 = blockIdx.y * 128 + wm * 32;
  int n0 = blockIdx.x * 128 + wn * 64;

  v8f zero = {};
  v8f acc[2][4];
  #pragma unroll
  for (int i = 0; i < 2; ++i)
    #pragma unroll
    for (int j = 0; j < 4; ++j) acc[i][j] = zero;

  for (int k0 = 0; k0 < K; k0 += 32) {
    v16bf a0 = load_a_frag(A, K, m0,      k0, lane);
    v16bf a1 = load_a_frag(A, K, m0 + 16, k0, lane);
    #pragma unroll
    for (int j = 0; j < 4; ++j) {
      v16bf bj = load_b_frag(B, K, n0 + 16 * j, k0, lane);
      acc[0][j] = wmma_bf16(a0, bj, acc[0][j]);
      acc[1][j] = wmma_bf16(a1, bj, acc[1][j]);
    }
  }

  int hi = lane >> 4, nn = lane & 15;
  #pragma unroll
  for (int i = 0; i < 2; ++i) {
    #pragma unroll
    for (int j = 0; j < 4; ++j) {
      int col = n0 + 16 * j + nn;
      float bv = bias ? bias[col] : 0.0f;
      #pragma unroll
      for (int r = 0; r < 8; ++r) {
        int row = m0 + 16 * i + r + hi * 8;
        float v = acc[i][j][r] + bv;
        if (Cf) Cf[(size_t)row * N + col] = v;
        else    Cb[(size_t)row * N + col] = f2bf(v);
      }
    }
  }
}

// ---------- L2-normalize q,k (in place) ; scatter V^T ----------

__global__ __launch_bounds__(256) void norm_qkv(
    bf16_t* __restrict__ q, bf16_t* __restrict__ k,
    const bf16_t* __restrict__ v, bf16_t* __restrict__ vt,
    const float* __restrict__ scale_p) {
  int lane = threadIdx.x & 31;
  int wave = threadIdx.x >> 5;
  int rid = blockIdx.x * 8 + wave;            // (b*SLEN+s)*NHEADS + h
  int h  = rid & (NHEADS - 1);
  int ms = rid >> 4;                          // b*SLEN + s
  int b  = ms / SLEN, s = ms - b * SLEN;
  size_t base = (size_t)ms * DIMN + (size_t)h * DHEAD + lane * 4;

  const unsigned short* qp = (const unsigned short*)q;
  const unsigned short* kp = (const unsigned short*)k;
  const unsigned short* vp = (const unsigned short*)v;

  ushort4 qr = *(const ushort4*)(qp + base);
  ushort4 kr = *(const ushort4*)(kp + base);
  ushort4 vr = *(const ushort4*)(vp + base);

  float q0 = bf2f(qr.x), q1 = bf2f(qr.y), q2 = bf2f(qr.z), q3 = bf2f(qr.w);
  float k0 = bf2f(kr.x), k1 = bf2f(kr.y), k2 = bf2f(kr.z), k3 = bf2f(kr.w);

  float qs = q0 * q0 + q1 * q1 + q2 * q2 + q3 * q3;
  float ks = k0 * k0 + k1 * k1 + k2 * k2 + k3 * k3;
  #pragma unroll
  for (int off = 16; off >= 1; off >>= 1) {
    qs += __shfl_xor(qs, off, 32);
    ks += __shfl_xor(ks, off, 32);
  }
  float qmul = rsqrtf(qs) * scale_p[0];
  float kmul = rsqrtf(ks);

  ushort4 qo, ko;
  qo.x = f2bfu(q0 * qmul); qo.y = f2bfu(q1 * qmul);
  qo.z = f2bfu(q2 * qmul); qo.w = f2bfu(q3 * qmul);
  ko.x = f2bfu(k0 * kmul); ko.y = f2bfu(k1 * kmul);
  ko.z = f2bfu(k2 * kmul); ko.w = f2bfu(k3 * kmul);
  *(ushort4*)((unsigned short*)q + base) = qo;
  *(ushort4*)((unsigned short*)k + base) = ko;

  unsigned short* vtp = (unsigned short*)vt;
  size_t vtb = ((size_t)(b * NHEADS + h) * DHEAD + lane * 4) * SLEN + s;
  vtp[vtb + 0 * SLEN] = vr.x;
  vtp[vtb + 1 * SLEN] = vr.y;
  vtp[vtb + 2 * SLEN] = vr.z;
  vtp[vtb + 3 * SLEN] = vr.w;
}

// ---------- causal flash attention with async double-buffered K/V in LDS ----------
// grid (SLEN/128, BSZ*NHEADS); 8 waves/block; wave owns 16 q rows.
// All waves run the SAME key range [0, q0_blk+127]: fully-masked tiles are an
// exact no-op under -1e30 masking, which makes block barriers + shared staging legal.

__device__ __forceinline__ void stage_kv_tile(const bf16_t* __restrict__ Kb,
                                              const bf16_t* __restrict__ Vb,
                                              int key0, bf16_t* kdst, bf16_t* vdst,
                                              int tid) {
  #pragma unroll
  for (int i = 0; i < 2; ++i) {
    int c = tid + i * 256;
    int kr = c >> 4, ko = (c & 15) * 8;           // K tile: 32 rows x 128 dh
    async_copy16(Kb + (size_t)(key0 + kr) * DIMN + ko, kdst + kr * 128 + ko);
    int vr = c >> 2, vo = (c & 3) * 8;            // V^T tile: 128 dh x 32 keys
    async_copy16(Vb + (size_t)vr * SLEN + key0 + vo, vdst + vr * 32 + vo);
  }
}

__global__ __launch_bounds__(256) void flash_attn(
    const bf16_t* __restrict__ qn, const bf16_t* __restrict__ kn,
    const bf16_t* __restrict__ vt, bf16_t* __restrict__ ctx) {
  __shared__ bf16_t Kt[2][32 * 128];    // 16 KB
  __shared__ bf16_t Vt[2][128 * 32];    // 16 KB
  __shared__ bf16_t pbuf[8][16 * 32];   //  8 KB
  int tid = threadIdx.x;
  int lane = tid & 31;
  int wave = tid >> 5;
  int bh = blockIdx.y;
  int b = bh >> 4, h = bh & 15;
  int q0blk = blockIdx.x * 128;
  int q0 = q0blk + wave * 16;
  int nkt = (q0blk >> 5) + 4;           // key tiles covering [0, q0blk+127]

  const bf16_t* Qb = qn + (size_t)b * SLEN * DIMN + (size_t)h * DHEAD;
  const bf16_t* Kb = kn + (size_t)b * SLEN * DIMN + (size_t)h * DHEAD;
  const bf16_t* Vb = vt + (size_t)bh * DHEAD * SLEN;
  bf16_t* pmine = pbuf[wave];

  int hi = lane >> 4, nn = lane & 15;
  int qbase = q0 + hi * 8;

  v16bf qf[4];
  #pragma unroll
  for (int t = 0; t < 4; ++t) qf[t] = load_a_frag(Qb, DIMN, q0, t * 32, lane);

  v8f zero = {};
  v8f ctxacc[8];
  #pragma unroll
  for (int j = 0; j < 8; ++j) ctxacc[j] = zero;
  float mrow[8], lrow[8];
  #pragma unroll
  for (int r = 0; r < 8; ++r) { mrow[r] = -1e30f; lrow[r] = 0.0f; }

  stage_kv_tile(Kb, Vb, 0, Kt[0], Vt[0], tid);

  for (int kt = 0; kt < nkt; ++kt) {
    int key0 = kt * 32;
    int cur = kt & 1;
    if (kt + 1 < nkt) {
      stage_kv_tile(Kb, Vb, key0 + 32, Kt[cur ^ 1], Vt[cur ^ 1], tid);
      WAIT_ASYNC(4);                    // this tile's 4 async copies done; next 4 in flight
    } else {
      WAIT_ASYNC(0);
    }
    __syncthreads();

    // S tile: 16 q rows x 32 keys (two 16x16 accumulators), K from LDS
    v8f s0 = zero, s1 = zero;
    #pragma unroll
    for (int t = 0; t < 4; ++t) {
      v16bf kb0 = load_b_frag(Kt[cur], 128, 0,  t * 32, lane);
      v16bf kb1 = load_b_frag(Kt[cur], 128, 16, t * 32, lane);
      s0 = wmma_bf16(qf[t], kb0, s0);
      s1 = wmma_bf16(qf[t], kb1, s1);
    }

    // causal mask + per-row tile max (reduce across 16-lane C-layout groups)
    float tmax[8];
    #pragma unroll
    for (int r = 0; r < 8; ++r) {
      int qi = qbase + r;
      if (key0 + nn      > qi) s0[r] = -1e30f;
      if (key0 + 16 + nn > qi) s1[r] = -1e30f;
      float v = fmaxf(s0[r], s1[r]);
      v = fmaxf(v, __shfl_xor(v, 1, 32));
      v = fmaxf(v, __shfl_xor(v, 2, 32));
      v = fmaxf(v, __shfl_xor(v, 4, 32));
      v = fmaxf(v, __shfl_xor(v, 8, 32));
      tmax[r] = v;
    }

    // online softmax update
    float alpha[8];
    v8f p0, p1;
    #pragma unroll
    for (int r = 0; r < 8; ++r) {
      float nm = fmaxf(mrow[r], tmax[r]);
      alpha[r] = __expf(mrow[r] - nm);
      mrow[r] = nm;
      p0[r] = __expf(s0[r] - nm);
      p1[r] = __expf(s1[r] - nm);
      float ps = p0[r] + p1[r];
      ps += __shfl_xor(ps, 1, 32);
      ps += __shfl_xor(ps, 2, 32);
      ps += __shfl_xor(ps, 4, 32);
      ps += __shfl_xor(ps, 8, 32);
      lrow[r] = lrow[r] * alpha[r] + ps;
    }
    #pragma unroll
    for (int j = 0; j < 8; ++j) {
      #pragma unroll
      for (int r = 0; r < 8; ++r) ctxacc[j][r] *= alpha[r];
    }

    // repack P (C-layout) -> A-layout through per-wave LDS buffer
    #pragma unroll
    for (int r = 0; r < 8; ++r) {
      int ml = r + hi * 8;
      pmine[ml * 32 + nn]      = f2bf(p0[r]);
      pmine[ml * 32 + nn + 16] = f2bf(p1[r]);
    }
    asm volatile("s_wait_dscnt 0" ::: "memory");
    v16bf pa = load_a_frag(pmine, 32, 0, 0, lane);

    // ctx += P(16x32) @ V(32x128), V^T from LDS
    #pragma unroll
    for (int j = 0; j < 8; ++j) {
      v16bf vb = load_b_frag(Vt[cur], 32, j * 16, 0, lane);
      ctxacc[j] = wmma_bf16(pa, vb, ctxacc[j]);
    }
    __syncthreads();   // all waves done with buf[cur] before it is overwritten
  }

  #pragma unroll
  for (int r = 0; r < 8; ++r) lrow[r] = 1.0f / lrow[r];

  #pragma unroll
  for (int j = 0; j < 8; ++j) {
    #pragma unroll
    for (int r = 0; r < 8; ++r) {
      int row = q0 + r + hi * 8;
      int col = h * DHEAD + j * 16 + nn;
      ctx[(size_t)(b * SLEN + row) * DIMN + col] = f2bf(ctxacc[j][r] * lrow[r]);
    }
  }
}

// ---------- host ----------

extern "C" void kernel_launch(void* const* d_in, const int* in_sizes, int n_in,
                              void* d_out, int out_size, void* d_ws, size_t ws_size,
                              hipStream_t stream) {
  (void)in_sizes; (void)n_in; (void)out_size; (void)ws_size;
  const float* x     = (const float*)d_in[0];
  // d_in[1]: causal mask == tril(ones); applied analytically in flash_attn.
  const float* Wq    = (const float*)d_in[2];
  const float* bq    = (const float*)d_in[3];
  const float* Wk    = (const float*)d_in[4];
  const float* bk    = (const float*)d_in[5];
  const float* Wv    = (const float*)d_in[6];
  const float* bv    = (const float*)d_in[7];
  const float* Wo    = (const float*)d_in[8];
  const float* bo    = (const float*)d_in[9];
  const float* scale = (const float*)d_in[10];
  float* out = (float*)d_out;

  const size_t XE = (size_t)MTOT * DIMN;
  const size_t WE = (size_t)DIMN * DIMN;

  uint8_t* w = (uint8_t*)d_ws;
  bf16_t* xb  = (bf16_t*)w; w += XE * 2;
  bf16_t* wqb = (bf16_t*)w; w += WE * 2;
  bf16_t* wkb = (bf16_t*)w; w += WE * 2;
  bf16_t* wvb = (bf16_t*)w; w += WE * 2;
  bf16_t* wob = (bf16_t*)w; w += WE * 2;
  bf16_t* qrw = (bf16_t*)w; w += XE * 2;
  bf16_t* krw = (bf16_t*)w; w += XE * 2;
  bf16_t* vrw = (bf16_t*)w; w += XE * 2;
  bf16_t* vtb = (bf16_t*)w; w += XE * 2;   // V^T [bh, dh, slen]
  bf16_t* ctb = (bf16_t*)w; w += XE * 2;

  dim3 blk(256);

  cvt_f32_bf16<<<dim3((unsigned)(XE / 4 / 256)), blk, 0, stream>>>(x,  xb,  (int)XE);
  cvt_f32_bf16<<<dim3((unsigned)(WE / 4 / 256)), blk, 0, stream>>>(Wq, wqb, (int)WE);
  cvt_f32_bf16<<<dim3((unsigned)(WE / 4 / 256)), blk, 0, stream>>>(Wk, wkb, (int)WE);
  cvt_f32_bf16<<<dim3((unsigned)(WE / 4 / 256)), blk, 0, stream>>>(Wv, wvb, (int)WE);
  cvt_f32_bf16<<<dim3((unsigned)(WE / 4 / 256)), blk, 0, stream>>>(Wo, wob, (int)WE);

  dim3 gg(DIMN / 128, MTOT / 128);         // (16, 32)
  gemm_bf16_nt<<<gg, blk, 0, stream>>>(xb, wqb, bq, nullptr, qrw, MTOT, DIMN, DIMN);
  gemm_bf16_nt<<<gg, blk, 0, stream>>>(xb, wkb, bk, nullptr, krw, MTOT, DIMN, DIMN);
  gemm_bf16_nt<<<gg, blk, 0, stream>>>(xb, wvb, bv, nullptr, vrw, MTOT, DIMN, DIMN);

  norm_qkv<<<dim3((unsigned)(BSZ * SLEN * NHEADS / 8)), blk, 0, stream>>>(
      qrw, krw, vrw, vtb, scale);

  dim3 fg(SLEN / 128, BSZ * NHEADS);       // (16, 32)
  flash_attn<<<fg, blk, 0, stream>>>(qrw, krw, vtb, ctb);

  gemm_bf16_nt<<<gg, blk, 0, stream>>>(ctb, wob, bo, out, nullptr, MTOT, DIMN, DIMN);
}
